// AttnPooling_2138893713779
// MI455X (gfx1250) — matrix operand order repaired and checked by reading, not compiled, not run on hardware
//
#include <hip/hip_runtime.h>
#include <hip/hip_bf16.h>
#include <hip/hip_fp16.h>

// ---------------------------------------------------------------------------
// AttnPooling on MI455X (gfx1250):
//   prep  : split w1 into (f16 hi, f16 lo) pairs pre-packed in WMMA B-fragment
//           per-lane layout (done once; 32 KB)
//   pass 1: s = tanh(H @ w1^T) @ w2 via v_wmma_f32_16x16x32_f16, split-fp16
//           (hi*hi + hi*lo + lo*hi) for ~fp32 accuracy; hardware v_tanh_f32;
//           ds_swizzle xor-butterfly for the w2 dot reduction
//   pass 2: per-segment softmax + weighted segment-sum of H (batch sorted ->
//           one block per segment; deterministic, no atomics)
// ---------------------------------------------------------------------------

typedef __attribute__((ext_vector_type(16))) _Float16 v16h;
typedef __attribute__((ext_vector_type(8)))  float    v8f;
typedef __attribute__((ext_vector_type(8)))  unsigned v8u;

#define IN_DIM       128
#define HID_DIM      64
#define TILE_NODES   128
#define GEMM_THREADS 256

__device__ __forceinline__ float fast_tanh(float x) {
#if __has_builtin(__builtin_amdgcn_tanhf)
  return __builtin_amdgcn_tanhf(x);      // V_TANH_F32 (gfx1250 trans op)
#else
  return tanhf(x);
#endif
}

// xor-butterfly add within each 16-lane half (masks 1,2,4,8 stay in-group)
template <int MASK>
__device__ __forceinline__ float swz_add(float v) {
#if __has_builtin(__builtin_amdgcn_ds_swizzle)
  // group-of-32 swizzle: lane' = ((lane & 0x1f) | 0) ^ MASK
  int t = __builtin_amdgcn_ds_swizzle(__builtin_bit_cast(int, v),
                                      (MASK << 10) | 0x1f);
  return v + __builtin_bit_cast(float, t);
#else
  return v + __shfl_xor(v, MASK, 32);
#endif
}
__device__ __forceinline__ float half_group_sum(float v) {
  v = swz_add<1>(v);
  v = swz_add<2>(v);
  v = swz_add<4>(v);
  v = swz_add<8>(v);
  return v;
}

__device__ __forceinline__ unsigned short h_bits(_Float16 h) {
  return __builtin_bit_cast(unsigned short, h);
}
// pack two f16 into a dword: low half = first element (matches v16h layout)
__device__ __forceinline__ unsigned pack_pair(_Float16 a, _Float16 b) {
  return (unsigned)h_bits(a) | ((unsigned)h_bits(b) << 16);
}

// ---------------------------------------------------------------------------
// Prep: w1 [64x128] f32 -> split hi/lo f16 fragments in per-lane B layout.
// dword index ((nt*4+ks)*32 + lane)*16 + r      : b_hi dword r
//             ((nt*4+ks)*32 + lane)*16 + 8 + r  : b_lo dword r
// B fragment (32x16 f16): lane n = lane&15, half = lane>>4,
//   k = ks*32 + half*16 + 2r (+1);  B[k][n] = w1[nt*16+n][k]
// ---------------------------------------------------------------------------
__global__ __launch_bounds__(256)
void attnpool_prep_w1(const float* __restrict__ w1, unsigned* __restrict__ w1frag)
{
  const int t = threadIdx.x;
  #pragma unroll
  for (int i = 0; i < 16; ++i) {
    int idx  = t + i * 256;            // 0..4095 == (nt, ks, lane, r)
    int r    = idx & 7;
    int lane = (idx >> 3) & 31;
    int ks   = (idx >> 8) & 3;
    int nt   = (idx >> 10) & 3;
    int half = lane >> 4, nrow = lane & 15;
    int hcol = nt * 16 + nrow;
    int k    = ks * 32 + (half << 4) + (r << 1);
    float x0 = w1[hcol * IN_DIM + k];
    float x1 = w1[hcol * IN_DIM + k + 1];
    _Float16 h0 = (_Float16)x0, h1 = (_Float16)x1;
    _Float16 l0 = (_Float16)(x0 - (float)h0);
    _Float16 l1 = (_Float16)(x1 - (float)h1);
    int base = ((nt * 4 + ks) * 32 + lane) << 4;
    w1frag[base + r]     = pack_pair(h0, h1);
    w1frag[base + 8 + r] = pack_pair(l0, l1);
  }
}

// ---------------------------------------------------------------------------
// Kernel 1: per-node score s[n] = sum_h w2[h] * tanh(sum_k H[n,k]*w1[h,k])
// 8 waves/block, each wave owns a 16-node stripe of a 128-node tile.
// ks-outer / nt-inner: A fragment loaded once per ks, reused by 4 hid tiles.
// ---------------------------------------------------------------------------
__global__ __launch_bounds__(GEMM_THREADS)
void attnpool_gemm_score(const float* __restrict__ H,
                         const unsigned* __restrict__ w1frag,
                         const float* __restrict__ w2,
                         float* __restrict__ s,
                         int n_nodes)
{
  // H tile as pair-packed f16 planes: Hhi[node][d] = hi(k=2d)|hi(2d+1)<<16
  // Swizzle XORs only bits [5:2] of d -> aligned groups of 4 dwords stay
  // contiguous, so fragment reads are ds_load_b128.
  __shared__ unsigned Hhi[TILE_NODES][IN_DIM / 2];   // 32 KB
  __shared__ unsigned Hlo[TILE_NODES][IN_DIM / 2];   // 32 KB

  const int tid   = threadIdx.x;
  const int tile0 = blockIdx.x * TILE_NODES;

  // ---- stage tile: coalesced float2 loads, split+pack into two planes ----
  const float2* H2 = (const float2*)H;
  #pragma unroll
  for (int i = 0; i < (TILE_NODES * IN_DIM / 2) / GEMM_THREADS; ++i) {
    int pair = tid + i * GEMM_THREADS;       // 0..8191
    int node = pair >> 6;                    // 64 pairs per node
    int dpos = pair & 63;
    int g    = tile0 + node;
    float2 x = (g < n_nodes) ? H2[(size_t)g * (IN_DIM / 2) + dpos]
                             : make_float2(0.f, 0.f);
    _Float16 h0 = (_Float16)x.x, h1 = (_Float16)x.y;
    _Float16 l0 = (_Float16)(x.x - (float)h0);
    _Float16 l1 = (_Float16)(x.y - (float)h1);
    int sw = (node & 15) << 2;               // bank-spread swizzle (bits 5:2)
    Hhi[node][dpos ^ sw] = pack_pair(h0, h1);
    Hlo[node][dpos ^ sw] = pack_pair(l0, l1);
  }
  __syncthreads();

  const int wave  = tid >> 5;
  const int lane  = tid & 31;
  const int half  = lane >> 4;
  const int nrow  = lane & 15;
  const int anode = wave * 16 + nrow;
  const int asw   = (anode & 15) << 2;

  v8f accs[4];
  #pragma unroll
  for (int nt = 0; nt < 4; ++nt) accs[nt] = {};

  for (int ks = 0; ks < IN_DIM / 32; ++ks) {
    // A fragment (16x32 f16): dword r covers k = ks*32+16*half+2*r
    // -> d = k/2; r groups 0..3 / 4..7 are 4 consecutive dwords each.
    int d0 = (ks * 16 + (half << 3)) ^ asw;        // r = 0..3
    int d1 = (ks * 16 + (half << 3) + 4) ^ asw;    // r = 4..7
    uint4 a0 = *(const uint4*)&Hhi[anode][d0];
    uint4 a1 = *(const uint4*)&Hhi[anode][d1];
    uint4 l0 = *(const uint4*)&Hlo[anode][d0];
    uint4 l1 = *(const uint4*)&Hlo[anode][d1];
    v8u au, lu;
    au[0]=a0.x; au[1]=a0.y; au[2]=a0.z; au[3]=a0.w;
    au[4]=a1.x; au[5]=a1.y; au[6]=a1.z; au[7]=a1.w;
    lu[0]=l0.x; lu[1]=l0.y; lu[2]=l0.z; lu[3]=l0.w;
    lu[4]=l1.x; lu[5]=l1.y; lu[6]=l1.z; lu[7]=l1.w;
    v16h a_hi = __builtin_bit_cast(v16h, au);
    v16h a_lo = __builtin_bit_cast(v16h, lu);

    #pragma unroll
    for (int nt = 0; nt < 4; ++nt) {
      const unsigned* bp = w1frag + (((nt * 4 + ks) * 32 + lane) << 4);
      v16h b_hi = __builtin_bit_cast(v16h, *(const v8u*)(bp));
      v16h b_lo = __builtin_bit_cast(v16h, *(const v8u*)(bp + 8));

      // split-precision accumulate: hi*hi + hi*lo + lo*hi
      accs[nt] = __builtin_amdgcn_wmma_f32_16x16x32_f16(false, a_hi, false, b_hi,
                                                        (short)0, accs[nt], false, false);
      accs[nt] = __builtin_amdgcn_wmma_f32_16x16x32_f16(false, a_hi, false, b_lo,
                                                        (short)0, accs[nt], false, false);
      accs[nt] = __builtin_amdgcn_wmma_f32_16x16x32_f16(false, a_lo, false, b_hi,
                                                        (short)0, accs[nt], false, false);
    }
  }

  // Epilogue: acc[j] = x[m][n], m = 8*half + j, n = nrow (C/D layout).
  // s[m] = sum_h w2[h]*tanh(x[m][h]) = sum_nt sum_n : accumulate per-lane over
  // nt first, then one 16-lane butterfly per row j.
  float s_acc[8];
  #pragma unroll
  for (int j = 0; j < 8; ++j) s_acc[j] = 0.0f;

  #pragma unroll
  for (int nt = 0; nt < 4; ++nt) {
    float w2v = w2[nt * 16 + nrow];
    #pragma unroll
    for (int j = 0; j < 8; ++j)
      s_acc[j] = fmaf(fast_tanh(accs[nt][j]), w2v, s_acc[j]);
  }
  #pragma unroll
  for (int j = 0; j < 8; ++j) s_acc[j] = half_group_sum(s_acc[j]);

  // Lane (half*16 + j), j in 0..7, writes node m = half*8 + j of this wave.
  int jj = lane & 15;
  if (jj < 8) {
    float v = s_acc[0];
    #pragma unroll
    for (int j = 1; j < 8; ++j) if (jj == j) v = s_acc[j];
    int gn = tile0 + wave * 16 + half * 8 + jj;
    if (gn < n_nodes) s[gn] = v;
  }
}

// ---------------------------------------------------------------------------
// Kernel 2: one block (128 threads = IN_DIM) per segment; batch is sorted, so
// binary-search the [beg,end) node range -> deterministic reductions.
// ---------------------------------------------------------------------------
__global__ __launch_bounds__(IN_DIM)
void attnpool_segment_pool(const float* __restrict__ H,
                           const long long* __restrict__ batch,
                           const float* __restrict__ s,
                           float* __restrict__ out,
                           int n_nodes)
{
  __shared__ int   bounds[2];
  __shared__ float red[IN_DIM];
  __shared__ float ebuf[IN_DIM];

  const int g   = blockIdx.x;
  const int tid = threadIdx.x;

  if (tid == 0) {
    long long vg = (long long)g;
    int lo = 0, hi = n_nodes;
    while (lo < hi) { int mid = (lo + hi) >> 1; if (batch[mid] < vg) lo = mid + 1; else hi = mid; }
    bounds[0] = lo;
    int lo2 = lo, hi2 = n_nodes;
    long long vg1 = vg + 1;
    while (lo2 < hi2) { int mid = (lo2 + hi2) >> 1; if (batch[mid] < vg1) lo2 = mid + 1; else hi2 = mid; }
    bounds[1] = lo2;
  }
  __syncthreads();
  const int beg = bounds[0], end = bounds[1];

  // segment max
  float mloc = -INFINITY;
  for (int i = beg + tid; i < end; i += IN_DIM) mloc = fmaxf(mloc, s[i]);
  red[tid] = mloc; __syncthreads();
  for (int st = IN_DIM / 2; st > 0; st >>= 1) {
    if (tid < st) red[tid] = fmaxf(red[tid], red[tid + st]);
    __syncthreads();
  }
  float m = red[0];
  if (beg == end) m = 0.0f;               // isfinite guard (empty segment)
  __syncthreads();

  // segment sum of exp(s - m)
  float dloc = 0.0f;
  for (int i = beg + tid; i < end; i += IN_DIM) dloc += __expf(s[i] - m);
  red[tid] = dloc; __syncthreads();
  for (int st = IN_DIM / 2; st > 0; st >>= 1) {
    if (tid < st) red[tid] += red[tid + st];
    __syncthreads();
  }
  float denom = red[0];
  __syncthreads();
  float inv = (beg < end) ? 1.0f / denom : 0.0f;

  // acc[f] = sum_n softmax_n * H[n, f]; thread == feature -> coalesced rows
  float acc = 0.0f;
  for (int base = beg; base < end; base += IN_DIM) {
    int cnt = min(IN_DIM, end - base);
    if (tid < cnt) ebuf[tid] = __expf(s[base + tid] - m) * inv;
    __syncthreads();
    const float* Hrow = H + (size_t)base * IN_DIM + tid;
    for (int j = 0; j < cnt; ++j)
      acc += ebuf[j] * Hrow[(size_t)j * IN_DIM];
    __syncthreads();
  }
  out[(size_t)g * IN_DIM + tid] = acc;
}

// ---------------------------------------------------------------------------
extern "C" void kernel_launch(void* const* d_in, const int* in_sizes, int n_in,
                              void* d_out, int out_size, void* d_ws, size_t ws_size,
                              hipStream_t stream) {
  const float*     H     = (const float*)d_in[0];      // [N, 128] f32
  const float*     w1    = (const float*)d_in[1];      // [64, 128] f32
  const float*     w2    = (const float*)d_in[2];      // [1, 64]  f32
  const long long* batch = (const long long*)d_in[3];  // [N] int64, sorted

  const int n_nodes = in_sizes[0] / IN_DIM;
  const int n_seg   = out_size / IN_DIM;               // G (out is [G, 128])

  // workspace layout: [ s : N floats ][ w1frag : 8192 dwords ]
  float*    s_scratch = (float*)d_ws;
  size_t    off       = (((size_t)n_nodes * sizeof(float)) + 15) & ~(size_t)15;
  unsigned* w1frag    = (unsigned*)((char*)d_ws + off);
  float*    out       = (float*)d_out;

  attnpool_prep_w1<<<1, 256, 0, stream>>>(w1, w1frag);
  const int nblk = (n_nodes + TILE_NODES - 1) / TILE_NODES;
  attnpool_gemm_score<<<nblk, GEMM_THREADS, 0, stream>>>(H, w1frag, w2, s_scratch, n_nodes);
  attnpool_segment_pool<<<n_seg, IN_DIM, 0, stream>>>(H, batch, s_scratch, out, n_nodes);
}
